// PaperGCNModel_87617332838752
// MI455X (gfx1250) — matrix operand order repaired and checked by reading, not compiled, not run on hardware
//
#include <hip/hip_runtime.h>
#include <hip/hip_bf16.h>

// ---------------- problem constants (match reference) ----------------
#define NN   100000
#define EE   3200000
#define CC   256
#define HH   64
#define GG   512
#define EPSV 1e-5f

typedef __attribute__((ext_vector_type(16))) __bf16 v16bf;
typedef __attribute__((ext_vector_type(8)))  __bf16 v8bf;
typedef __attribute__((ext_vector_type(8)))  float  v8f;
typedef int v4i __attribute__((vector_size(16)));   // matches builtin's expected pointee

#if __has_builtin(__builtin_amdgcn_global_load_async_to_lds_b128) && \
    __has_builtin(__builtin_amdgcn_s_wait_asynccnt)
#define USE_ASYNC_LDS 1
#endif

// ---------------- utility kernels ----------------
__global__ void k_fill(float* __restrict__ p, float v, long long n) {
    long long i = (long long)blockIdx.x * blockDim.x + threadIdx.x;
    if (i < n) p[i] = v;
}

__global__ void k_cast_bf16(const float* __restrict__ in, __bf16* __restrict__ out, long long n) {
    long long i = (long long)blockIdx.x * blockDim.x + threadIdx.x;
    if (i < n) out[i] = (__bf16)in[i];
}

// Pre-swizzle W [KDIM x 64] (row-major f32) into per-lane WMMA B-fragment order (bf16):
// Bp[((ct*(KDIM/32)+kb32)*32 + lane)*16 + e] = W[(kb32*32 + (lane>>4)*16 + e)*64 + ct*16 + (lane&15)]
__global__ void k_pack_b(const float* __restrict__ W, __bf16* __restrict__ Bp, int KDIM) {
    int idx = blockIdx.x * blockDim.x + threadIdx.x;
    if (idx >= KDIM * 64) return;
    int e    = idx & 15;
    int lane = (idx >> 4) & 31;
    int rest = idx >> 9;                 // ct*(KDIM/32) + kb32
    int nkb  = KDIM / 32;
    int kb32 = rest % nkb;
    int ct   = rest / nkb;
    int half = lane >> 4, ml = lane & 15;
    int k = kb32 * 32 + half * 16 + e;
    Bp[idx] = (__bf16)W[k * 64 + ct * 16 + ml];
}

// deg starts at 1.0 (self loop); each edge adds 1 at dst
__global__ void k_deg(const int* __restrict__ dst, float* __restrict__ deg) {
    long long e = (long long)blockIdx.x * blockDim.x + threadIdx.x;
    if (e < EE) atomicAdd(&deg[dst[e]], 1.0f);
}

__global__ void k_rsqrt_inplace(float* __restrict__ deg) {
    long long i = (long long)blockIdx.x * blockDim.x + threadIdx.x;
    if (i < NN) deg[i] = rsqrtf(deg[i]);   // deg >= 1 always (self loops)
}

// ---------------- bf16 WMMA GEMM: [nrows x KDIM] @ [KDIM x 64] -> f32 [nrows x 64]
// block = 128 threads (4 waves); each wave owns one 16-col tile; block covers 64 cols.
// Packed B for the whole block is staged in LDS (async-to-LDS on CDNA5), then
// RTILES row-tiles of 16 rows are processed against it.
template<int KDIM, int RTILES>
__global__ __launch_bounds__(128)
void k_gemm_bf16_n64(const __bf16* __restrict__ A, const __bf16* __restrict__ Bp,
                     float* __restrict__ C, int ntiles) {
    __shared__ __bf16 __attribute__((aligned(32))) sB[KDIM * 64]; // pre-swizzled fragments
    const int wave = threadIdx.x >> 5;
    const int lane = threadIdx.x & 31;
    const int half = lane >> 4;
    const int ml   = lane & 15;

    // --- stage packed B (KDIM*128 bytes): 128 threads x (KDIM/16) x 16B chunks ---
    constexpr int CHUNKS = KDIM / 16;
#ifdef USE_ASYNC_LDS
    #pragma unroll
    for (int c = 0; c < CHUNKS; ++c) {
        int off = (c * 128 + threadIdx.x) * 16;    // bytes
        __builtin_amdgcn_global_load_async_to_lds_b128(
            (__attribute__((address_space(1))) v4i*)((const char*)Bp + off),
            (__attribute__((address_space(3))) v4i*)((char*)sB + off), 0, 0);
    }
    __builtin_amdgcn_s_wait_asynccnt(0);
#else
    #pragma unroll
    for (int c = 0; c < CHUNKS; ++c) {
        int eo = (c * 128 + threadIdx.x) * 8;      // bf16 elements (16B)
        *(v8bf*)(sB + eo) = *(const v8bf*)(Bp + eo);
    }
#endif
    __syncthreads();

    const __bf16* bbase = sB + (long long)wave * KDIM * 16;  // this wave's column tile

    for (int t = 0; t < RTILES; ++t) {
        long long tile = (long long)blockIdx.x * RTILES + t;
        if (tile >= ntiles) break;
        const long long r0 = tile * 16;
        const __bf16* arow = A + (r0 + ml) * (long long)KDIM;

        v8f acc = {};
        #pragma unroll
        for (int kb32 = 0; kb32 < KDIM / 32; ++kb32) {
            // A fragment: lanes 0-15 K=kb+0..15; lanes 16-31 K=kb+16..31
            v8bf alo = *(const v8bf*)(arow + kb32 * 32 + half * 16);
            v8bf ahi = *(const v8bf*)(arow + kb32 * 32 + half * 16 + 8);
            v16bf a = __builtin_shufflevector(alo, ahi, 0, 1, 2, 3, 4, 5, 6, 7,
                                              8, 9, 10, 11, 12, 13, 14, 15);
            // B fragment: one contiguous 32B LDS read in lane-fragment order
            v16bf b = *(const v16bf*)(bbase + (kb32 * 32 + lane) * 16);
            acc = __builtin_amdgcn_wmma_f32_16x16x32_bf16(false, a, false, b,
                                                          (short)0, acc, false, false);
        }
        // C/D layout: VGPR r holds M = r + half*8, N = ml
        float* crow = C + (r0 + half * 8) * 64 + wave * 16 + ml;
        #pragma unroll
        for (int r = 0; r < 8; ++r) crow[(long long)r * 64] = acc[r];
    }
}

// ---------------- GCN aggregation ----------------
// agg[i][f] = dis[i]^2 * h[i][f] + bias[f]   (self-loop term + bias)
__global__ void k_agg_init(const float* __restrict__ h, const float* __restrict__ dis,
                           const float* __restrict__ bias, float* __restrict__ agg) {
    long long idx = (long long)blockIdx.x * blockDim.x + threadIdx.x;
    if (idx >= (long long)NN * 64) return;
    long long i = idx >> 6; int f = idx & 63;
    float d = dis[i];
    agg[idx] = d * d * h[idx] + bias[f];
}

// one 64-thread group per edge; norm computed once per edge via LDS
__global__ __launch_bounds__(256)
void k_agg_edges(const float* __restrict__ h, const float* __restrict__ dis,
                 const int* __restrict__ src, const int* __restrict__ dst,
                 float* __restrict__ agg) {
    __shared__ float snorm[4];
    __shared__ int   ssrc[4], sdst[4];
    const int le = threadIdx.x >> 6;   // 0..3 local edge
    const int f  = threadIdx.x & 63;   // feature
    const long long e = (long long)blockIdx.x * 4 + le;   // EE % 4 == 0
    if (f == 0) {
        int s = src[e], d = dst[e];
        ssrc[le] = s; sdst[le] = d;
        snorm[le] = dis[s] * dis[d];
    }
    __syncthreads();
    float val = snorm[le] * h[(long long)ssrc[le] * 64 + f];
    atomicAdd(&agg[(long long)sdst[le] * 64 + f], val);
}

// ---------------- BatchNorm (training-mode batch stats) over relu(agg) ----------------
__global__ __launch_bounds__(256)
void k_bn_stats(const float* __restrict__ agg, float* __restrict__ stats) {
    __shared__ float ssum[256], ssq[256];
    const int f  = threadIdx.x & 63;
    const int rg = threadIdx.x >> 6;   // 0..3
    float s = 0.f, q = 0.f;
    for (long long i = (long long)blockIdx.x * 4 + rg; i < NN; i += (long long)gridDim.x * 4) {
        float v = fmaxf(agg[i * 64 + f], 0.f);
        s += v; q += v * v;
    }
    ssum[threadIdx.x] = s; ssq[threadIdx.x] = q;
    __syncthreads();
    if (threadIdx.x < 64) {
        s = ssum[f] + ssum[64 + f] + ssum[128 + f] + ssum[192 + f];
        q = ssq[f] + ssq[64 + f] + ssq[128 + f] + ssq[192 + f];
        atomicAdd(&stats[f], s);
        atomicAdd(&stats[64 + f], q);
    }
}

__global__ void k_bn_apply(const float* __restrict__ agg, const float* __restrict__ stats,
                           const float* __restrict__ g, const float* __restrict__ be,
                           float* __restrict__ hnorm, __bf16* __restrict__ hbf) {
    long long idx = (long long)blockIdx.x * blockDim.x + threadIdx.x;
    if (idx >= (long long)NN * 64) return;
    int f = idx & 63;
    const float invn = 1.0f / (float)NN;
    float m   = stats[f] * invn;
    float var = stats[64 + f] * invn - m * m;
    float v = fmaxf(agg[idx], 0.f);
    float o = (v - m) * rsqrtf(var + EPSV) * g[f] + be[f];
    hnorm[idx] = o;
    hbf[idx] = (__bf16)o;
}

// ---------------- global mean pool ----------------
__global__ void k_pool(const float* __restrict__ h, const int* __restrict__ batch,
                       float* __restrict__ pool, float* __restrict__ cnt) {
    long long idx = (long long)blockIdx.x * blockDim.x + threadIdx.x;
    if (idx >= (long long)NN * 64) return;
    long long i = idx >> 6; int f = idx & 63;
    int gidx = batch[i];
    atomicAdd(&pool[(long long)gidx * 64 + f], h[idx]);
    if (f == 0) atomicAdd(&cnt[gidx], 1.0f);
}

__global__ void k_pool_div(float* __restrict__ pool, const float* __restrict__ cnt) {
    int idx = blockIdx.x * blockDim.x + threadIdx.x;
    if (idx >= GG * 64) return;
    pool[idx] /= fmaxf(cnt[idx >> 6], 1.0f);
}

// ---------------- small dense layers (MLP head) ----------------
__global__ void k_dense(const float* __restrict__ X, const float* __restrict__ W,
                        const float* __restrict__ bias, float* __restrict__ Y,
                        int M, int K, int Nout, int relu) {
    int idx = blockIdx.x * blockDim.x + threadIdx.x;
    if (idx >= M * Nout) return;
    int m = idx / Nout, n = idx % Nout;
    float s = bias[n];
    for (int k = 0; k < K; ++k) s += X[(long long)m * K + k] * W[(long long)k * Nout + n];
    Y[idx] = relu ? fmaxf(s, 0.f) : s;
}

// ---------------- launch ----------------
static inline int cdiv(long long a, long long b) { return (int)((a + b - 1) / b); }

extern "C" void kernel_launch(void* const* d_in, const int* in_sizes, int n_in,
                              void* d_out, int out_size, void* d_ws, size_t ws_size,
                              hipStream_t stream) {
    (void)in_sizes; (void)n_in; (void)out_size; (void)ws_size;
    const float* x   = (const float*)d_in[0];
    const int*   ei  = (const int*)d_in[1];      // [2, E]: src then dst
    const int*   bat = (const int*)d_in[2];
    const float* w1  = (const float*)d_in[3];
    const float* b1  = (const float*)d_in[4];
    const float* w2  = (const float*)d_in[5];
    const float* b2  = (const float*)d_in[6];
    const float* g1  = (const float*)d_in[7];
    const float* be1 = (const float*)d_in[8];
    const float* g2  = (const float*)d_in[9];
    const float* be2 = (const float*)d_in[10];
    const float* fw1 = (const float*)d_in[11];
    const float* fb1 = (const float*)d_in[12];
    const float* fw2 = (const float*)d_in[13];
    const float* fb2 = (const float*)d_in[14];
    const float* fw3 = (const float*)d_in[15];
    const float* fb3 = (const float*)d_in[16];
    const float* fw4 = (const float*)d_in[17];
    const float* fb4 = (const float*)d_in[18];
    const float* ow  = (const float*)d_in[19];
    const float* ob  = (const float*)d_in[20];
    float* out = (float*)d_out;

    char* ws = (char*)d_ws;
    // workspace layout (bytes, 16B-aligned)
    float*  deg   = (float*)(ws + 0);                      // N f32 -> becomes dis after rsqrt
    __bf16* xbf   = (__bf16*)(ws + 400128);                // N*C bf16
    __bf16* w1bf  = (__bf16*)(ws + 51600128);              // C*H bf16 (packed fragments)
    __bf16* w2bf  = (__bf16*)(ws + 51632896);              // H*H bf16 (packed fragments)
    float*  bufA  = (float*)(ws + 51641088);               // N*64 f32
    float*  bufB  = (float*)(ws + 77241088);               // N*64 f32
    __bf16* hbf   = (__bf16*)(ws + 102841088);             // N*64 bf16
    float*  stats = (float*)(ws + 115641088);              // 128 f32 (sum, sumsq)
    float*  pool  = (float*)(ws + 115641600);              // G*64 f32
    float*  cnt   = (float*)(ws + 115772672);              // G f32 (contiguous after pool)
    float*  m0    = (float*)(ws + 115774720);              // G*128 f32
    float*  m1    = (float*)(ws + 116036864);              // G*128 f32

    const int* esrc = ei;
    const int* edst = ei + EE;
    const long long NH = (long long)NN * 64;
    const int NTILES = NN / 16;            // 6250
    const int RT = 8;                      // row-tiles per block

    // --- precision conversion + weight fragment packing ---
    k_cast_bf16<<<cdiv((long long)NN * CC, 256), 256, 0, stream>>>(x, xbf, (long long)NN * CC);
    k_pack_b<<<cdiv(CC * 64, 256), 256, 0, stream>>>(w1, w1bf, CC);
    k_pack_b<<<cdiv(HH * 64, 256), 256, 0, stream>>>(w2, w2bf, HH);

    // --- degree / symmetric normalization ---
    k_fill<<<cdiv(NN, 256), 256, 0, stream>>>(deg, 1.0f, NN);          // self loops
    k_deg<<<cdiv(EE, 256), 256, 0, stream>>>(edst, deg);
    k_rsqrt_inplace<<<cdiv(NN, 256), 256, 0, stream>>>(deg);           // deg -> dis

    // --- GCN layer 1: h = x @ w1 (WMMA bf16), aggregate, relu+BN ---
    k_gemm_bf16_n64<CC, 8><<<cdiv(NTILES, RT), 128, 0, stream>>>(xbf, w1bf, bufA, NTILES);
    k_agg_init<<<cdiv(NH, 256), 256, 0, stream>>>(bufA, deg, b1, bufB);
    k_agg_edges<<<EE / 4, 256, 0, stream>>>(bufA, deg, esrc, edst, bufB);
    k_fill<<<1, 128, 0, stream>>>(stats, 0.0f, 128);
    k_bn_stats<<<512, 256, 0, stream>>>(bufB, stats);
    k_bn_apply<<<cdiv(NH, 256), 256, 0, stream>>>(bufB, stats, g1, be1, bufA, hbf);

    // --- GCN layer 2: h = hbf @ w2 (WMMA bf16), aggregate, relu+BN ---
    k_gemm_bf16_n64<HH, 8><<<cdiv(NTILES, RT), 128, 0, stream>>>(hbf, w2bf, bufB, NTILES);
    k_agg_init<<<cdiv(NH, 256), 256, 0, stream>>>(bufB, deg, b2, bufA);
    k_agg_edges<<<EE / 4, 256, 0, stream>>>(bufB, deg, esrc, edst, bufA);
    k_fill<<<1, 128, 0, stream>>>(stats, 0.0f, 128);
    k_bn_stats<<<512, 256, 0, stream>>>(bufA, stats);
    k_bn_apply<<<cdiv(NH, 256), 256, 0, stream>>>(bufA, stats, g2, be2, bufB, hbf);

    // --- global mean pool ---
    k_fill<<<cdiv(GG * 64 + GG, 256), 256, 0, stream>>>(pool, 0.0f, GG * 64 + GG); // pool + cnt
    k_pool<<<cdiv(NH, 256), 256, 0, stream>>>(bufB, bat, pool, cnt);
    k_pool_div<<<cdiv(GG * 64, 256), 256, 0, stream>>>(pool, cnt);

    // --- MLP head ---
    k_dense<<<cdiv(GG * 128, 256), 256, 0, stream>>>(pool, fw1, fb1, m0, GG, 64, 128, 1);
    k_dense<<<cdiv(GG * 64, 256), 256, 0, stream>>>(m0, fw2, fb2, m1, GG, 128, 64, 1);
    k_dense<<<cdiv(GG * 32, 256), 256, 0, stream>>>(m1, fw3, fb3, m0, GG, 64, 32, 1);
    k_dense<<<cdiv(GG * 16, 256), 256, 0, stream>>>(m0, fw4, fb4, m1, GG, 32, 16, 1);
    k_dense<<<cdiv(GG * 2, 256), 256, 0, stream>>>(m1, ow, ob, out, GG, 16, 2, 0);
}
